// MRFLoss_4329327035149
// MI455X (gfx1250) — matrix operand order repaired and checked by reading, not compiled, not run on hardware
//
#include <hip/hip_runtime.h>
#include <hip/hip_bf16.h>

typedef __attribute__((ext_vector_type(16))) _Float16 v16h;
typedef __attribute__((ext_vector_type(8)))  _Float16 v8h;
typedef __attribute__((ext_vector_type(8)))  float    v8f;

#define MRF_EPS 1e-5f

// ---------------------------------------------------------------------------
// A operand tile load: 16x32 (MxK) f16, row-major rows of length ldK.
// ISA layout (16-bit A 16x32): lanes 0-15 -> M=lane, K={0..7}U{16..23};
// lanes 16-31 -> M=lane-16, K={8..15}U{24..31}.
// ---------------------------------------------------------------------------
__device__ __forceinline__ v16h load_tile_A(const _Float16* tile, int ldK) {
  const int lane = threadIdx.x & 31;
  const int row  = lane & 15;
  const int kofs = (lane >> 4) << 3;          // 0 or 8
  const _Float16* p = tile + (size_t)row * ldK + kofs;
  v8h lo = *(const v8h*)(p);                  // K kofs..kofs+7
  v8h hi = *(const v8h*)(p + 16);             // K kofs+16..kofs+23
  v16h r;
#pragma unroll
  for (int i = 0; i < 8; ++i) { r[i] = lo[i]; r[i + 8] = hi[i]; }
  return r;
}

// ---------------------------------------------------------------------------
// B operand tile load: 32x16 (KxN) f16 where B[k][q] = G[q][k], G row-major.
// ISA layout (16-bit B): lanes 0-15 -> N=lane, K={0..7}U{16..23}; lanes 16-31 -> N=lane-16,
// K={8..15}U{24..31} (K packed 2 per VGPR).
// ---------------------------------------------------------------------------
__device__ __forceinline__ v16h load_tile_B(const _Float16* tile, int ldK) {
  const int lane = threadIdx.x & 31;
  const int col  = lane & 15;
  const int kofs = (lane >> 4) << 3;          // 0 or 8
  const _Float16* p = tile + (size_t)col * ldK + kofs;
  v8h lo = *(const v8h*)(p);
  v8h hi = *(const v8h*)(p + 16);
  v16h r;
#pragma unroll
  for (int i = 0; i < 8; ++i) { r[i] = lo[i]; r[i + 8] = hi[i]; }
  return r;
}

// 16x16 f32 tile of cd-dot = sum_c T[p][c]*G[q][c], K-loop over C in steps of 32
__device__ __forceinline__ v8f gemm_tile(const _Float16* Trow, const _Float16* Grow, int C) {
  v8f acc = {};
  for (int k0 = 0; k0 < C; k0 += 32) {
    v16h a = load_tile_A(Trow + k0, C);
    v16h b = load_tile_B(Grow + k0, C);
    acc = __builtin_amdgcn_wmma_f32_16x16x32_f16(
        /*neg_a=*/false, a, /*neg_b=*/false, b,
        /*c_mod=*/(short)0, acc, /*reuse_a=*/false, /*reuse_b=*/false);
  }
  return acc;
}

// ---------------------------------------------------------------------------
// Preprocess: subtract target channel-mean, L2-normalize over C, emit f16
// operand matrices: T[n][p][c] (A) and G[n][q][c] (B source, q == gen pixel).
// One thread per (n, pixel); three strided passes over C (HBM-light: 38 MB).
// ---------------------------------------------------------------------------
__global__ void mrf_prep_kernel(const float* __restrict__ gen,
                                const float* __restrict__ tar,
                                _Float16* __restrict__ T,
                                _Float16* __restrict__ G,
                                int C, int P, int total) {
  int gidx = blockIdx.x * blockDim.x + threadIdx.x;
  if (gidx >= total) return;
  int n = gidx / P;
  int p = gidx - n * P;
  const float* tb = tar + (size_t)n * C * P + p;
  const float* gb = gen + (size_t)n * C * P + p;

  float s = 0.f;
  for (int c = 0; c < C; ++c) s += tb[(size_t)c * P];
  float m = s / (float)C;

  float st = 0.f, sg = 0.f;
  for (int c = 0; c < C; ++c) {
    float tv = tb[(size_t)c * P] - m;
    float gv = gb[(size_t)c * P] - m;
    st += tv * tv;
    sg += gv * gv;
  }
  float ti = rsqrtf(st);
  float gi = rsqrtf(sg);

  _Float16* Trow = T + (size_t)gidx * C;
  _Float16* Grow = G + (size_t)gidx * C;
  for (int c = 0; c < C; ++c) {
    Trow[c] = (_Float16)((tb[(size_t)c * P] - m) * ti);
    Grow[c] = (_Float16)((gb[(size_t)c * P] - m) * gi);
  }
}

// ---------------------------------------------------------------------------
// Stripe kernel: block = (n, stripe of 4 q-tiles = 64 gen columns).
// 8 waves = 2 p-phases x 4 q-tiles. Three WMMA sweeps over all p-tiles:
//  1) column min of cd
//  2) column sum Z of e = exp(2*(1 - cd/(min+eps)))
//  3) row max of e/Z  -> global atomicMax (int bits, values > 0)
// ---------------------------------------------------------------------------
__global__ void mrf_stripe_kernel(const _Float16* __restrict__ T,
                                  const _Float16* __restrict__ G,
                                  float* __restrict__ rowmax,
                                  int C, int P, int numPTiles, int numQTiles) {
  __shared__ float s_cmin[2][64];
  __shared__ float s_zsum[2][64];

  const int lane = threadIdx.x & 31;
  const int w    = threadIdx.x >> 5;   // 0..7
  const int qt   = w & 3;              // q-tile within stripe
  const int ph   = w >> 2;             // p phase 0/1
  const int n    = blockIdx.y;
  const int qTile = blockIdx.x * 4 + qt;
  const bool qvalid = (qTile < numQTiles);   // uniform per wave

  const _Float16* Tn   = T + (size_t)n * P * C;
  const _Float16* Grow = G + (size_t)n * P * C + (size_t)qTile * 16 * C;

  // ---- sweep 1: column min --------------------------------------------
  float denom = 1.0f;
  if (qvalid) {
    float cmin = 3.0e38f;
    for (int pt = ph; pt < numPTiles; pt += 2) {
      v8f acc = gemm_tile(Tn + (size_t)pt * 16 * C, Grow, C);
#pragma unroll
      for (int i = 0; i < 8; ++i) cmin = fminf(cmin, 0.5f - 0.5f * acc[i]);
    }
    cmin = fminf(cmin, __shfl_xor(cmin, 16, 32));  // combine M halves
    if (lane < 16) s_cmin[ph][qt * 16 + lane] = cmin;
  }
  __syncthreads();

  // ---- sweep 2: column Z ----------------------------------------------
  if (qvalid) {
    denom = fminf(s_cmin[0][qt * 16 + (lane & 15)],
                  s_cmin[1][qt * 16 + (lane & 15)]) + MRF_EPS;
    float z = 0.f;
    for (int pt = ph; pt < numPTiles; pt += 2) {
      v8f acc = gemm_tile(Tn + (size_t)pt * 16 * C, Grow, C);
#pragma unroll
      for (int i = 0; i < 8; ++i) {
        float cd = 0.5f - 0.5f * acc[i];
        z += __expf(2.0f * (1.0f - cd / denom));
      }
    }
    z += __shfl_xor(z, 16, 32);
    if (lane < 16) s_zsum[ph][qt * 16 + lane] = z;
  }
  __syncthreads();

  // ---- sweep 3: row max of cs = e/Z -----------------------------------
  if (qvalid) {
    float invZ = 1.0f / (s_zsum[0][qt * 16 + (lane & 15)] +
                         s_zsum[1][qt * 16 + (lane & 15)]);
    float* rmn = rowmax + (size_t)n * P;
    for (int pt = ph; pt < numPTiles; pt += 2) {
      v8f acc = gemm_tile(Tn + (size_t)pt * 16 * C, Grow, C);
      float r[8];
#pragma unroll
      for (int i = 0; i < 8; ++i) {
        float cd = 0.5f - 0.5f * acc[i];
        r[i] = __expf(2.0f * (1.0f - cd / denom)) * invZ;
      }
      // butterfly max across the 16 lanes of each half (over the 16 q cols)
#pragma unroll
      for (int msk = 1; msk < 16; msk <<= 1) {
#pragma unroll
        for (int i = 0; i < 8; ++i) r[i] = fmaxf(r[i], __shfl_xor(r[i], msk, 32));
      }
      if ((lane & 15) == 0) {
        int rowbase = pt * 16 + ((lane >> 4) << 3);  // lanes<16: M=i, else M=8+i
#pragma unroll
        for (int i = 0; i < 8; ++i)
          atomicMax((int*)&rmn[rowbase + i], __float_as_int(r[i]));
      }
    }
  }
}

// ---------------------------------------------------------------------------
// Finalize: per sample, loss_n = -log(mean_p rowmax[n][p]); out += w * loss_n
// ---------------------------------------------------------------------------
__global__ void mrf_finalize_kernel(const float* __restrict__ rowmax,
                                    int P, float weight, float* __restrict__ out) {
  __shared__ float red[256];
  const float* rmn = rowmax + (size_t)blockIdx.x * P;
  float s = 0.f;
  for (int i = threadIdx.x; i < P; i += 256) s += rmn[i];
  red[threadIdx.x] = s;
  __syncthreads();
  for (int o = 128; o > 0; o >>= 1) {
    if (threadIdx.x < o) red[threadIdx.x] += red[threadIdx.x + o];
    __syncthreads();
  }
  if (threadIdx.x == 0) {
    float mean = red[0] / (float)P;
    atomicAdd(out, weight * (-__logf(mean)));
  }
}

extern "C" void kernel_launch(void* const* d_in, const int* in_sizes, int n_in,
                              void* d_out, int out_size, void* d_ws, size_t ws_size,
                              hipStream_t stream) {
  const float* gen3 = (const float*)d_in[0];
  const float* tar3 = (const float*)d_in[1];
  const float* gen4 = (const float*)d_in[2];
  const float* tar4 = (const float*)d_in[3];
  float* out = (float*)d_out;

  const int N = 4;
  const int C3 = 256, P3 = 56 * 56;   // 3136, 196 tiles
  const int C4 = 512, P4 = 28 * 28;   // 784, 49 tiles

  char* ws = (char*)d_ws;
  size_t off = 0;
  auto wsalloc = [&](size_t bytes) -> void* {
    void* p = ws + off;
    off = (off + bytes + 255) & ~(size_t)255;
    return p;
  };
  _Float16* T3 = (_Float16*)wsalloc((size_t)N * P3 * C3 * sizeof(_Float16));
  _Float16* G3 = (_Float16*)wsalloc((size_t)N * P3 * C3 * sizeof(_Float16));
  _Float16* T4 = (_Float16*)wsalloc((size_t)N * P4 * C4 * sizeof(_Float16));
  _Float16* G4 = (_Float16*)wsalloc((size_t)N * P4 * C4 * sizeof(_Float16));
  float* rm3 = (float*)wsalloc((size_t)N * P3 * sizeof(float));
  float* rm4 = (float*)wsalloc((size_t)N * P4 * sizeof(float));

  hipMemsetAsync(rm3, 0, (size_t)N * P3 * sizeof(float), stream);
  hipMemsetAsync(rm4, 0, (size_t)N * P4 * sizeof(float), stream);
  hipMemsetAsync(out, 0, (size_t)out_size * sizeof(float), stream);

  mrf_prep_kernel<<<(N * P3 + 255) / 256, 256, 0, stream>>>(gen3, tar3, T3, G3, C3, P3, N * P3);
  mrf_prep_kernel<<<(N * P4 + 255) / 256, 256, 0, stream>>>(gen4, tar4, T4, G4, C4, P4, N * P4);

  const int qT3 = P3 / 16;                      // 196
  const int qT4 = P4 / 16;                      // 49
  dim3 g3((qT3 + 3) / 4, N);
  dim3 g4((qT4 + 3) / 4, N);
  mrf_stripe_kernel<<<g3, 256, 0, stream>>>(T3, G3, rm3, C3, P3, P3 / 16, qT3);
  mrf_stripe_kernel<<<g4, 256, 0, stream>>>(T4, G4, rm4, C4, P4, P4 / 16, qT4);

  mrf_finalize_kernel<<<N, 256, 0, stream>>>(rm3, P3, 0.5f, out);
  mrf_finalize_kernel<<<N, 256, 0, stream>>>(rm4, P4, 1.0f, out);
}
